// MinGRUBlock_49581102465607
// MI455X (gfx1250) — compile-verified, matrix operand order checked by reading, not run on hardware
//
#include <hip/hip_runtime.h>
#include <math.h>

// ---------------------------------------------------------------------------
// MinGRU block for MI455X (gfx1250).
// GEMMs: bf16 WMMA, 128x128x32 tiles, weights pre-transposed to N-major so
// BOTH tiles stream via global_load_async_to_lds_b128 (ASYNCcnt, double
// buffered).  All N dims padded to multiples of 128 -> no guards anywhere.
// ---------------------------------------------------------------------------

typedef __attribute__((ext_vector_type(16))) __bf16 v16bf;
typedef __attribute__((ext_vector_type(8)))  float  v8f;

#define Bz   4
#define Lw   4096
#define Dd   1024
#define Hh   4
#define HDd  256
#define DIi  384
#define FFI  2730          // int(1024*4*2/3)
#define FFN  2816          // FFI padded to multiple of 128 (N/K pad for WMMA)
#define Tt   (Bz*Lw)       // 16384 tokens

__device__ __forceinline__ float  bf2f(__bf16 v){ return (float)v; }
__device__ __forceinline__ __bf16 f2bf(float v){ return (__bf16)v; }

// 32-byte per-lane async copy global -> LDS (same INST_OFFSET applies to both
// the LDS and global address per CDNA5 ISA 10.7/15.18).
__device__ __forceinline__ void async_copy32(void* lds, const void* gptr)
{
    unsigned l = (unsigned)(unsigned long long)lds;   // flat[31:0] == LDS addr
    asm volatile("global_load_async_to_lds_b128 %0, %1, off\n\t"
                 "global_load_async_to_lds_b128 %0, %1, off offset:16"
                 :: "v"(l), "v"(gptr) : "memory");
}
__device__ __forceinline__ void wait_async0()
{
    asm volatile("s_wait_asynccnt 0x0" ::: "memory");
}

// --------------------------- rmsnorm (f32 -> bf16) -------------------------
__global__ __launch_bounds__(256)
void k_rmsnorm(const float* __restrict__ x, const float* __restrict__ gamma,
               __bf16* __restrict__ out)
{
    const int t = blockIdx.x, tid = threadIdx.x;
    const float4* xr = (const float4*)(x + (size_t)t * Dd);
    float4 v = xr[tid];                       // 256 threads * 4 = 1024
    float ss = v.x*v.x + v.y*v.y + v.z*v.z + v.w*v.w;
#pragma unroll
    for (int o = 16; o; o >>= 1) ss += __shfl_down(ss, o, 32);
    __shared__ float red[8];
    __shared__ float s_inv;
    if ((tid & 31) == 0) red[tid >> 5] = ss;
    __syncthreads();
    if (tid == 0) {
        float s = red[0]+red[1]+red[2]+red[3]+red[4]+red[5]+red[6]+red[7];
        s_inv = 32.0f / fmaxf(sqrtf(s), 1e-12f);   // sqrt(1024) = 32
    }
    __syncthreads();
    const float inv = s_inv;
    float4 g = ((const float4*)gamma)[tid];
    __bf16* o = out + (size_t)t * Dd + tid * 4;
    o[0] = f2bf(v.x * inv * (g.x + 1.0f));
    o[1] = f2bf(v.y * inv * (g.y + 1.0f));
    o[2] = f2bf(v.z * inv * (g.z + 1.0f));
    o[3] = f2bf(v.w * inv * (g.w + 1.0f));
}

// --------------------- causal depthwise conv, K=4 --------------------------
__global__ __launch_bounds__(256)
void k_dwconv(const __bf16* __restrict__ xn, const float* __restrict__ w,
              const float* __restrict__ b, __bf16* __restrict__ out)
{
    const int i  = blockIdx.x * 256 + threadIdx.x;   // over Tt*Dd
    const int d  = i & (Dd - 1);
    const int tk = i >> 10;
    const int l  = tk & (Lw - 1);
    float acc = b[d];
#pragma unroll
    for (int k = 0; k < 4; ++k) {
        int ls = l - 3 + k;
        if (ls >= 0)
            acc += bf2f(xn[(size_t)(tk - 3 + k) * Dd + d]) * w[d * 4 + k];
    }
    out[i] = f2bf(acc);
}

// ----------------------- weight conversion helpers -------------------------
__global__ __launch_bounds__(256)
void k_cvt(const float* __restrict__ in, __bf16* __restrict__ out, int n)
{
    int i = blockIdx.x * 256 + threadIdx.x;
    if (i < n) out[i] = f2bf(in[i]);
}

// Transpose-convert K x N (f32, row-major) -> Np x Kp (bf16, N-major), zero
// padding outside [Kin,Nin).  Batched over blockIdx.y.
__global__ __launch_bounds__(256)
void k_tr_cvt(const float* __restrict__ in, __bf16* __restrict__ out,
              int Kin, int Nin, int Kp, long long sIn, long long sOut,
              int perBatch)
{
    int i = blockIdx.x * 256 + threadIdx.x;
    if (i >= perBatch) return;
    int n = i / Kp, k = i % Kp;
    const float* inb = in + (size_t)blockIdx.y * sIn;
    float v = (k < Kin && n < Nin) ? inb[(size_t)k * Nin + n] : 0.0f;
    out[(size_t)blockIdx.y * sOut + i] = f2bf(v);
}

// ----------------------------- WMMA GEMM -----------------------------------
// C[M x N] = A[M x K](bf16 row-major) * B^T where Bn is N x K (bf16, N-major).
// M, N multiples of 128; K multiple of 32.  256 threads = 8 wave32; each wave
// owns a 32x64 sub-tile = 2x4 v_wmma_f32_16x16x32_bf16 accumulators.
// Tiles stream via async-to-LDS with double buffering.
__global__ __launch_bounds__(256)
void k_gemm(const __bf16* __restrict__ A, int lda, long long sA,
            const __bf16* __restrict__ Bn, int ldb, long long sB,
            void* __restrict__ Cv, int ldc, long long sC, int c_is_bf16,
            int K)
{
    __shared__ __bf16 As[2][128][48];   // [buf][m][k], 96B row stride
    __shared__ __bf16 Bs[2][128][48];   // [buf][n][k]

    const int tid   = threadIdx.x;
    const int tileM = blockIdx.y * 128;
    const int tileN = blockIdx.x * 128;
    const int z     = blockIdx.z;
    A  += (size_t)z * sA;
    Bn += (size_t)z * sB;

    const int wave = tid >> 5, lane = tid & 31;
    const int wm = wave & 3, wn = wave >> 2;      // 4 x 2 wave grid
    const int half = lane >> 4, r = lane & 15;

    const int row = tid >> 1, seg = tid & 1;      // 128 rows x 2 x 16 elems

    const __bf16* aBase = A  + (size_t)(tileM + row) * lda + seg * 16;
    const __bf16* bBase = Bn + (size_t)(tileN + row) * ldb + seg * 16;

    v8f acc[2][4] = {};

    const int nk = K >> 5;
    // prologue: DMA tile 0
    async_copy32(&As[0][row][seg * 16], aBase);
    async_copy32(&Bs[0][row][seg * 16], bBase);
    wait_async0();
    __syncthreads();

    for (int kt = 0; kt < nk; ++kt) {
        const int cur = kt & 1, nxt = cur ^ 1;
        if (kt + 1 < nk) {                        // DMA next tile while computing
            async_copy32(&As[nxt][row][seg * 16], aBase + (kt + 1) * 32);
            async_copy32(&Bs[nxt][row][seg * 16], bBase + (kt + 1) * 32);
        }

        v16bf af[2], bfv[4];
#pragma unroll
        for (int mt = 0; mt < 2; ++mt)
            af[mt]  = *(const v16bf*)&As[cur][wm * 32 + mt * 16 + r][half * 16];
#pragma unroll
        for (int nt = 0; nt < 4; ++nt)
            bfv[nt] = *(const v16bf*)&Bs[cur][wn * 64 + nt * 16 + r][half * 16];
#pragma unroll
        for (int mt = 0; mt < 2; ++mt)
#pragma unroll
            for (int nt = 0; nt < 4; ++nt)
                acc[mt][nt] = __builtin_amdgcn_wmma_f32_16x16x32_bf16(
                    false, af[mt], false, bfv[nt], (short)0, acc[mt][nt],
                    false, false);

        wait_async0();        // my DMAs into buf[nxt] have landed
        __syncthreads();      // everyone's DMAs landed / reads of buf[cur] done
    }

    // store C (16x16 f32 layout: vgpr j, lanes 0-15 -> M=j, lanes 16-31 -> M=j+8)
#pragma unroll
    for (int mt = 0; mt < 2; ++mt) {
#pragma unroll
        for (int nt = 0; nt < 4; ++nt) {
            int colb = tileN + wn * 64 + nt * 16 + r;
            int rowb = tileM + wm * 32 + mt * 16 + half * 8;
            if (c_is_bf16) {
                __bf16* C = (__bf16*)Cv + (size_t)z * sC;
#pragma unroll
                for (int j = 0; j < 8; ++j)
                    C[(size_t)(rowb + j) * ldc + colb] = f2bf(acc[mt][nt][j]);
            } else {
                float* C = (float*)Cv + (size_t)z * sC;
#pragma unroll
                for (int j = 0; j < 8; ++j)
                    C[(size_t)(rowb + j) * ldc + colb] = acc[mt][nt][j];
            }
        }
    }
}

// ------------------------ log-space minGRU scan ----------------------------
__device__ __forceinline__ float softplusf(float x)
{
    if (x >  20.0f) return x;
    if (x < -20.0f) return expf(x);
    return log1pf(expf(x));
}

// hg: (T, H*2*DI) bf16; hbuf: (T, H*DI) bf16; nh: (B, H*DI) f32
__global__ __launch_bounds__(256)
void k_scan(const __bf16* __restrict__ hg, __bf16* __restrict__ hbuf,
            float* __restrict__ nh)
{
    const int ch = blockIdx.x * 256 + threadIdx.x;   // 0 .. B*H*DI-1 (6144)
    const int b  = ch / (Hh * DIi);
    const int c  = ch % (Hh * DIi);
    const int h  = c / DIi, e = c % DIi;
    const size_t hidOff  = (size_t)h * 2 * DIi + e;
    const size_t gateOff = hidOff + DIi;
    float a = 0.0f, m = -3.0e38f, s = 0.0f, hval = 0.0f;
    for (int l = 0; l < Lw; ++l) {
        size_t rowO = ((size_t)b * Lw + l) * (Hh * 2 * DIi);
        float hid = bf2f(hg[rowO + hidOff]);
        float gt  = bf2f(hg[rowO + gateOff]);
        float lc  = -softplusf(gt);                        // log coeff
        float lg  = (hid >= 0.0f) ? logf(fmaxf(hid, 0.0f) + 0.5f)
                                  : -softplusf(-hid);      // _log_g
        float lv  = -softplusf(-gt) + lg;                  // log value
        a += lc;                                           // a_star
        float zv = lv - a;
        if (zv > m) { s = s * expf(m - zv) + 1.0f; m = zv; }
        else        { s += expf(zv - m); }
        hval = expf(a + m + logf(s));                      // exp(log_h)
        hbuf[((size_t)b * Lw + l) * (Hh * DIi) + c] = f2bf(hval);
    }
    nh[(size_t)b * (Hh * DIi) + c] = hval;                 // next_hidden
}

// ------------------------------ elementwise --------------------------------
__global__ __launch_bounds__(256)
void k_silu(const __bf16* __restrict__ g, const __bf16* __restrict__ v,
            __bf16* __restrict__ out)
{
    int i = blockIdx.x * 256 + threadIdx.x;   // over Tt*FFN
    float gf = bf2f(g[i]);                    // pad cols are 0 -> silu(0)*0 = 0
    out[i] = f2bf(gf / (1.0f + expf(-gf)) * bf2f(v[i]));
}

__global__ __launch_bounds__(256)
void k_addbiasres(float* __restrict__ C, const float* __restrict__ bias,
                  const float* __restrict__ res)
{
    int i = blockIdx.x * 256 + threadIdx.x;
    C[i] = C[i] + bias[i & (Dd - 1)] + res[i];
}

__global__ __launch_bounds__(256)
void k_addinp(float* __restrict__ x, const float* __restrict__ y)
{
    int i = blockIdx.x * 256 + threadIdx.x;
    x[i] += y[i];
}

// ------------------------------- launcher ----------------------------------
extern "C" void kernel_launch(void* const* d_in, const int* in_sizes, int n_in,
                              void* d_out, int out_size, void* d_ws, size_t ws_size,
                              hipStream_t stream)
{
    (void)in_sizes; (void)n_in; (void)out_size; (void)ws_size;
    const float* x        = (const float*)d_in[0];
    const float* dw_w     = (const float*)d_in[1];
    const float* dw_b     = (const float*)d_in[2];
    const float* pw_w     = (const float*)d_in[3];
    const float* pw_b     = (const float*)d_in[4];
    const float* conv_g   = (const float*)d_in[5];
    const float* gru_g    = (const float*)d_in[6];
    const float* ff_g     = (const float*)d_in[7];
    const float* w_hg     = (const float*)d_in[8];
    const float* w_out    = (const float*)d_in[9];
    const float* w_gate   = (const float*)d_in[10];
    const float* w_value  = (const float*)d_in[11];
    const float* w_ff_out = (const float*)d_in[12];

    char* ws = (char*)d_ws;
    size_t off = 0;
    auto alloc = [&](size_t bytes) -> char* {
        char* p = ws + off;
        off += (bytes + 255) & ~(size_t)255;
        return p;
    };

    __bf16* xn_bf    = (__bf16*)alloc((size_t)Tt * Dd * 2);   // normed (reused 3x)
    __bf16* convd_bf = (__bf16*)alloc((size_t)Tt * Dd * 2);
    __bf16* pwN_bf   = (__bf16*)alloc((size_t)Dd * Dd * 2);   // pw_w already N x K
    float*  x1       = (float*) alloc((size_t)Tt * Dd * 4);   // residual stream
    __bf16* whgT_bf  = (__bf16*)alloc((size_t)Hh * 2 * DIi * HDd * 2);  // 768x256 per head
    __bf16* hg_bf    = (__bf16*)alloc((size_t)Tt * Hh * 2 * DIi * 2);
    __bf16* hbuf_bf  = (__bf16*)alloc((size_t)Tt * Hh * DIi * 2);
    __bf16* woutT_bf = (__bf16*)alloc((size_t)Hh * HDd * DIi * 2);      // 256x384 per head
    float*  gru_f32  = (float*) alloc((size_t)Tt * Dd * 4);
    __bf16* wgT_bf   = (__bf16*)alloc((size_t)FFN * Dd * 2);  // 2816x1024
    __bf16* wvT_bf   = (__bf16*)alloc((size_t)FFN * Dd * 2);
    __bf16* g_bf     = (__bf16*)alloc((size_t)Tt * FFN * 2);
    __bf16* v_bf     = (__bf16*)alloc((size_t)Tt * FFN * 2);
    __bf16* gated_bf = (__bf16*)alloc((size_t)Tt * FFN * 2);
    __bf16* wffoT_bf = (__bf16*)alloc((size_t)Dd * FFN * 2);  // 1024x2816

    float* outx = (float*)d_out;                     // (B,L,D)
    float* nh   = outx + (size_t)Tt * Dd;            // (B,1,H*DI)

    const int ED = (Tt * Dd) / 256;                  // 65536 blocks

    // 1) conv branch: rmsnorm -> depthwise -> pointwise GEMM -> +bias +res
    k_rmsnorm<<<dim3(Tt), 256, 0, stream>>>(x, conv_g, xn_bf);
    k_dwconv<<<dim3(ED), 256, 0, stream>>>(xn_bf, dw_w, dw_b, convd_bf);
    k_cvt<<<dim3((Dd * Dd) / 256), 256, 0, stream>>>(pw_w, pwN_bf, Dd * Dd);
    k_gemm<<<dim3(8, 128, 1), 256, 0, stream>>>(convd_bf, Dd, 0,
                                                pwN_bf, Dd, 0,
                                                x1, Dd, 0, 0, Dd);
    k_addbiasres<<<dim3(ED), 256, 0, stream>>>(x1, pw_b, x);

    // 2) GRU branch: rmsnorm -> per-head in-proj -> scan -> per-head out-proj
    k_rmsnorm<<<dim3(Tt), 256, 0, stream>>>(x1, gru_g, xn_bf);
    k_tr_cvt<<<dim3((2 * DIi * HDd) / 256, Hh), 256, 0, stream>>>(
        w_hg, whgT_bf, HDd, 2 * DIi, HDd,
        (long long)HDd * 2 * DIi, (long long)2 * DIi * HDd, 2 * DIi * HDd);
    k_gemm<<<dim3(6, 128, 4), 256, 0, stream>>>(
        xn_bf, Dd, HDd,                              // A: + h*256 per head
        whgT_bf, HDd, (long long)2 * DIi * HDd,      // B: (768 x 256) per head
        hg_bf, Hh * 2 * DIi, 2 * DIi, 1,             // C: bf16, + h*768
        HDd);
    k_scan<<<dim3((Bz * Hh * DIi) / 256), 256, 0, stream>>>(hg_bf, hbuf_bf, nh);
    k_tr_cvt<<<dim3((HDd * DIi) / 256, Hh), 256, 0, stream>>>(
        w_out, woutT_bf, DIi, HDd, DIi,
        (long long)DIi * HDd, (long long)HDd * DIi, HDd * DIi);
    k_gemm<<<dim3(2, 128, 4), 256, 0, stream>>>(
        hbuf_bf, Hh * DIi, DIi,
        woutT_bf, DIi, (long long)HDd * DIi,
        gru_f32, Dd, HDd, 0,
        DIi);
    k_addinp<<<dim3(ED), 256, 0, stream>>>(x1, gru_f32);   // x2

    // 3) FF branch: rmsnorm -> gate/value GEMMs -> SiLU gate -> out GEMM -> +res
    k_rmsnorm<<<dim3(Tt), 256, 0, stream>>>(x1, ff_g, xn_bf);
    k_tr_cvt<<<dim3((FFN * Dd) / 256, 1), 256, 0, stream>>>(
        w_gate, wgT_bf, Dd, FFI, Dd, 0, 0, FFN * Dd);
    k_tr_cvt<<<dim3((FFN * Dd) / 256, 1), 256, 0, stream>>>(
        w_value, wvT_bf, Dd, FFI, Dd, 0, 0, FFN * Dd);
    k_tr_cvt<<<dim3((Dd * FFN) / 256, 1), 256, 0, stream>>>(
        w_ff_out, wffoT_bf, FFI, Dd, FFN, 0, 0, Dd * FFN);
    k_gemm<<<dim3(22, 128, 1), 256, 0, stream>>>(xn_bf, Dd, 0,
                                                 wgT_bf, Dd, 0,
                                                 g_bf, FFN, 0, 1, Dd);
    k_gemm<<<dim3(22, 128, 1), 256, 0, stream>>>(xn_bf, Dd, 0,
                                                 wvT_bf, Dd, 0,
                                                 v_bf, FFN, 0, 1, Dd);
    k_silu<<<dim3((Tt * FFN) / 256), 256, 0, stream>>>(g_bf, v_bf, gated_bf);
    k_gemm<<<dim3(8, 128, 1), 256, 0, stream>>>(gated_bf, FFN, 0,
                                                wffoT_bf, FFN, 0,
                                                outx, Dd, 0, 0, FFN);
    k_addinp<<<dim3(ED), 256, 0, stream>>>(outx, x1);       // x3 = ff + x2
}
